// VisionTransformer_11132555231856
// MI455X (gfx1250) — matrix operand order repaired and checked by reading, not compiled, not run on hardware
//
#include <hip/hip_runtime.h>
#include <hip/hip_bf16.h>

// ---------------------------------------------------------------------------
// ViT-Base/16 forward (talking-heads attention), CDNA5 / gfx1250, wave32.
// All GEMMs: v_wmma_f32_16x16x32_bf16, f32 accumulate.
//  - Weights converted+transposed ONCE per launch to bf16 k-major [K][N]
//    ("B-ready"); B fragments are 16 strided bf16 loads (one per k row) per 16x16 tile.
//  - A operand bf16 in memory: two global_load_b128 per 16x32 fragment.
//  - Inner loop: 4 b128 A loads + 64 bf16 B loads + 8 WMMAs, zero conversion VALU, no LDS.
//  - K always a multiple of 32 (attention probs zero-padded to 256 cols;
//    K^T transposed with zero-padded token dim).
//  - GELU fused into fc1 epilogue; epilogues emit bf16 copies consumed as the
//    next GEMM's A operand; fp32 C output optional.
// ---------------------------------------------------------------------------

#define DMODEL 768
#define NHEAD  12
#define HDIM   64
#define DEPTH_ 12
#define PSZ    16
#define IMGSZ  224
#define GRID14 14
#define NPATCH 196           // 14*14
#define NTOK   197           // +cls
#define NTOKP  256           // padded attention width (multiple of 64)
#define NMP    (NTOK * NTOKP)
#define MLPD   3072
#define NCLS   1000
#define NCLSP  1024          // padded head width
#define BATCH  32
#define PPE    768           // 3*16*16
#define ATTN_SCALE 0.125f    // 64^-0.5

typedef __attribute__((ext_vector_type(16))) __bf16 v16bf;
typedef __attribute__((ext_vector_type(8)))  float  v8f;

__device__ __forceinline__ __bf16 f2bf(float f) {
  union { float f; unsigned u; } in; in.f = f;
  unsigned u = in.u;
  u += 0x7FFFu + ((u >> 16) & 1u);        // round-to-nearest-even
  union { unsigned short s; __bf16 b; } out;
  out.s = (unsigned short)(u >> 16);
  return out.b;
}

// ---------------------------------------------------------------------------
// Batched WMMA GEMM, both operands bf16, B pre-laid-out k-major:
//   v = alpha*sum_k A[z][m][k]*Wb[z][k][n] + bias[n] (+ Cadd[z][m][n])
//   if gelu: v = gelu(v);  if (C) C[z][m][n] = v;  if (Cb) Cb[z][m][n] = bf16(v)
// Requirements: K % 32 == 0, lda/ldb % 8 == 0, 16B-aligned bases,
//               Wb has >= K rows and n0+63 < ldb columns readable.
// Block: 128 threads (4 independent waves). Wave: rows [tmw,tmw+32) x cols
// [n0,n0+64) = 2x4 tiles of 16x16, 8 WMMAs per 32-wide K-step.
// ---------------------------------------------------------------------------
__global__ __launch_bounds__(128)
void k_gemm(const __bf16* __restrict__ A, int lda, long long sA1, long long sA2,
            const __bf16* __restrict__ Wb, int ldb, long long sW1, long long sW2,
            float* __restrict__ C, int ldc, long long sC1, long long sC2,
            __bf16* __restrict__ Cb,
            const float* __restrict__ bias, const float* __restrict__ Cadd,
            int M, int N, int K, float alpha, int zdiv, int gelu)
{
  const int z  = blockIdx.z;
  const int zb = z / zdiv, zh = z % zdiv;
  A  += zb * sA1 + zh * sA2;
  Wb += zb * sW1 + zh * sW2;
  if (C)  C  += zb * sC1 + zh * sC2;
  if (Cb) Cb += zb * sC1 + zh * sC2;
  const float* Cr = Cadd ? (Cadd + zb * sC1 + zh * sC2) : nullptr;

  const int lane = threadIdx.x & 31;
  const int wave = threadIdx.x >> 5;
  const int l16  = lane & 15, half = lane >> 4;
  const int n0   = blockIdx.x * 64;
  const int tmw  = (blockIdx.y * 4 + wave) * 32;
  if (tmw >= M) return;                       // waves independent: safe

  // branchless-clamped A row pointers (padded rows never stored)
  const __bf16* Arow[2];
#pragma unroll
  for (int mt = 0; mt < 2; ++mt) {
    int r = tmw + mt * 16 + l16;
    r = (r > M - 1) ? (M - 1) : r;
    Arow[mt] = A + (long long)r * lda;
  }
  const int kb = 8 * half;                    // B-fragment k base within the step
  const __bf16* Brow = Wb + (long long)kb * ldb + n0 + l16;
  const long long bstep = (long long)32 * ldb;

  v8f acc[2][4] = {};

  for (int kk = 0; kk < K; kk += 32) {
    // ---- A fragments: two contiguous 8-bf16 runs -> 2x b128 ----
    v16bf afr[2];
#pragma unroll
    for (int mt = 0; mt < 2; ++mt) {
      const __bf16* p = Arow[mt] + kk + 8 * half;
      union { uint4 u[2]; v16bf v; } au;
      au.u[0] = *(const uint4*)(p);          // k = kk+8h .. +7
      au.u[1] = *(const uint4*)(p + 16);     // k = kk+16+8h .. +7
      afr[mt] = au.v;
    }
    // ---- B fragments: column n, k = 8h+s / 16+8h+s (strided rows) ----
#pragma unroll
    for (int nt = 0; nt < 4; ++nt) {
      const __bf16* p = Brow + nt * 16;
      union { uint4 u[2]; v16bf v; } bu;
#pragma unroll
      for (int s = 0; s < 8; ++s) {
        bu.v[s]     = p[(long long)s * ldb];
        bu.v[8 + s] = p[(long long)(16 + s) * ldb];
      }
#pragma unroll
      for (int mt = 0; mt < 2; ++mt)
        acc[mt][nt] = __builtin_amdgcn_wmma_f32_16x16x32_bf16(
            false, afr[mt], false, bu.v, (short)0, acc[mt][nt], false, false);
    }
    Brow += bstep;
  }

  // ---- epilogue: n = n0 + nt*16 + l16 ; m = tmw + mt*16 + j + 8*half ----
#pragma unroll
  for (int nt = 0; nt < 4; ++nt) {
    const int n = n0 + nt * 16 + l16;
    if (n < N) {
      const float bv = bias ? bias[n] : 0.0f;
#pragma unroll
      for (int mt = 0; mt < 2; ++mt) {
#pragma unroll
        for (int j = 0; j < 8; ++j) {
          const int m = tmw + mt * 16 + j + 8 * half;
          if (m < M) {
            float v = acc[mt][nt][j] * alpha + bv;
            if (Cr) v += Cr[(long long)m * ldc + n];
            if (gelu) v = 0.5f * v * (1.0f + erff(v * 0.70710678118654752f));
            if (C)  C [(long long)m * ldc + n] = v;
            if (Cb) Cb[(long long)m * ldc + n] = f2bf(v);
          }
        }
      }
    }
  }
}

// ---------------------------------------------------------------------------
// Weight convert+transpose: out[z][c*ldo + r] = (r<R) ? bf16(in[z][r*Ck+c]) : 0
// in: fp32 [R,Ck] row-major; out: bf16 k-major [Ck][ldo], zero-padded rows.
// ---------------------------------------------------------------------------
__global__ void k_wt(const float* __restrict__ in, __bf16* __restrict__ out,
                     int R, int Ck, int ldo, long long inStride, long long outStride)
{
  in  += (long long)blockIdx.z * inStride;
  out += (long long)blockIdx.z * outStride;
  long long idx = (long long)blockIdx.x * blockDim.x + threadIdx.x;
  const long long total = (long long)Ck * ldo;
  if (idx >= total) return;
  int r = (int)(idx % ldo);
  int c = (int)(idx / ldo);
  out[idx] = (r < R) ? f2bf(in[(long long)r * Ck + c]) : (__bf16)f2bf(0.0f);
}

// ---------------------------------------------------------------------------
// K^T extraction: kt[b,h,d,m] = (m<197) ? qkv_bf[b*197+m][768 + h*64 + d] : 0
// (bf16 -> bf16 copy; produces zero-padded [64][256] B-ready tiles per head)
// ---------------------------------------------------------------------------
__global__ void k_kt(const __bf16* __restrict__ qkvb, __bf16* __restrict__ kt)
{
  long long idx = (long long)blockIdx.x * blockDim.x + threadIdx.x;
  const long long total = (long long)BATCH * NHEAD * HDIM * NTOKP;
  if (idx >= total) return;
  int m = (int)(idx % NTOKP);
  long long tmp = idx / NTOKP;
  int d = (int)(tmp % HDIM); tmp /= HDIM;
  int h = (int)(tmp % NHEAD);
  int b = (int)(tmp / NHEAD);
  __bf16 v = f2bf(0.0f);
  if (m < NTOK)
    v = qkvb[((long long)b * NTOK + m) * (3 * DMODEL) + DMODEL + h * HDIM + d];
  kt[idx] = v;
}

// ---------------------------------------------------------------------------
// Patch extraction: x[B,3,224,224] -> xp_bf[B*196, 768] bf16 (c,py,px fastest)
// ---------------------------------------------------------------------------
__global__ void k_patchify(const float* __restrict__ x, __bf16* __restrict__ xp)
{
  long long idx = (long long)blockIdx.x * blockDim.x + threadIdx.x;
  const long long total = (long long)BATCH * NPATCH * PPE;
  if (idx >= total) return;
  int col = (int)(idx % PPE);
  int p   = (int)((idx / PPE) % NPATCH);
  int b   = (int)(idx / ((long long)PPE * NPATCH));
  int c  = col >> 8;
  int r  = col & 255;
  int py = r >> 4, px = r & 15;
  int pr = p / GRID14, pc = p % GRID14;
  long long src = (((long long)b * 3 + c) * IMGSZ + (pr * PSZ + py)) * IMGSZ + (pc * PSZ + px);
  xp[idx] = f2bf(x[src]);
}

// ---------------------------------------------------------------------------
// t[b,n,:] = (n==0 ? cls : xe[b,n-1,:]) + pos[n,:]
// ---------------------------------------------------------------------------
__global__ void k_add_cls_pos(const float* __restrict__ xe, const float* __restrict__ cls,
                              const float* __restrict__ pos, float* __restrict__ t)
{
  long long idx = (long long)blockIdx.x * blockDim.x + threadIdx.x;
  const long long total = (long long)BATCH * NTOK * DMODEL;
  if (idx >= total) return;
  int d = (int)(idx % DMODEL);
  int n = (int)((idx / DMODEL) % NTOK);
  int b = (int)(idx / ((long long)DMODEL * NTOK));
  float v = (n == 0) ? cls[d] : xe[((long long)b * NPATCH + (n - 1)) * DMODEL + d];
  t[idx] = v + pos[(long long)n * DMODEL + d];
}

// ---------------------------------------------------------------------------
// Row LayerNorm over DMODEL -> bf16 output. One block per row.
// ---------------------------------------------------------------------------
__global__ void k_layernorm(const float* __restrict__ X, const float* __restrict__ g,
                            const float* __restrict__ b, __bf16* __restrict__ Y)
{
  const long long row = blockIdx.x;
  const float* x = X + row * DMODEL;
  __bf16*      y = Y + row * DMODEL;
  __shared__ float red[256];
  const int tid = threadIdx.x;

  float s = 0.0f;
  for (int i = tid; i < DMODEL; i += 256) s += x[i];
  red[tid] = s; __syncthreads();
  for (int off = 128; off > 0; off >>= 1) {
    if (tid < off) red[tid] += red[tid + off];
    __syncthreads();
  }
  const float mean = red[0] * (1.0f / DMODEL);
  __syncthreads();

  float v = 0.0f;
  for (int i = tid; i < DMODEL; i += 256) { float d = x[i] - mean; v += d * d; }
  red[tid] = v; __syncthreads();
  for (int off = 128; off > 0; off >>= 1) {
    if (tid < off) red[tid] += red[tid + off];
    __syncthreads();
  }
  const float rstd = rsqrtf(red[0] * (1.0f / DMODEL) + 1e-6f);

  for (int i = tid; i < DMODEL; i += 256)
    y[i] = f2bf((x[i] - mean) * rstd * g[i] + b[i]);
}

// ---------------------------------------------------------------------------
// Talking heads: s[g] = sum_h in[b,h,e] * w[g*H+h]; e over 197*256 (padded).
// ---------------------------------------------------------------------------
__global__ void k_mix_heads(const float* __restrict__ in, const float* __restrict__ w,
                            float* __restrict__ out, __bf16* __restrict__ outb)
{
  const long long nm = NMP;
  long long idx = (long long)blockIdx.x * blockDim.x + threadIdx.x;
  const long long total = (long long)BATCH * nm;
  if (idx >= total) return;
  long long b = idx / nm, e = idx % nm;
  const float* ip = in + b * NHEAD * nm + e;
  float vals[NHEAD];
#pragma unroll
  for (int h = 0; h < NHEAD; ++h) vals[h] = ip[(long long)h * nm];
#pragma unroll
  for (int g = 0; g < NHEAD; ++g) {
    float s = 0.0f;
#pragma unroll
    for (int h = 0; h < NHEAD; ++h) s += vals[h] * w[g * NHEAD + h];
    if (out)  out [b * NHEAD * nm + (long long)g * nm + e] = s;
    if (outb) outb[b * NHEAD * nm + (long long)g * nm + e] = f2bf(s);
  }
}

// ---------------------------------------------------------------------------
// Row softmax over first L of Lpad elements, in place; pad cols set to zero.
// ---------------------------------------------------------------------------
__global__ void k_softmax(float* __restrict__ X, int L, int Lpad)
{
  float* x = X + (long long)blockIdx.x * Lpad;
  __shared__ float red[256];
  const int tid = threadIdx.x;

  float m = -1e30f;
  for (int i = tid; i < L; i += 256) m = fmaxf(m, x[i]);
  red[tid] = m; __syncthreads();
  for (int off = 128; off > 0; off >>= 1) {
    if (tid < off) red[tid] = fmaxf(red[tid], red[tid + off]);
    __syncthreads();
  }
  m = red[0]; __syncthreads();

  float s = 0.0f;
  for (int i = tid; i < L; i += 256) { float e = __expf(x[i] - m); x[i] = e; s += e; }
  red[tid] = s; __syncthreads();
  for (int off = 128; off > 0; off >>= 1) {
    if (tid < off) red[tid] += red[tid + off];
    __syncthreads();
  }
  const float inv = 1.0f / red[0];
  for (int i = tid; i < L; i += 256) x[i] *= inv;
  for (int i = L + tid; i < Lpad; i += 256) x[i] = 0.0f;   // zero the padding
}

// ---------------------------------------------------------------------------
// Host side
// ---------------------------------------------------------------------------
static void launch_gemm(hipStream_t s,
                        const __bf16* A, int lda, long long sA1, long long sA2,
                        const __bf16* Wb, int ldb, long long sW1, long long sW2,
                        float* C, int ldc, long long sC1, long long sC2,
                        __bf16* Cb, const float* bias, const float* Cadd,
                        int M, int N, int K, float alpha, int batches, int zdiv, int gelu)
{
  dim3 grid((N + 63) / 64, (M + 127) / 128, batches);
  k_gemm<<<grid, 128, 0, s>>>(A, lda, sA1, sA2, Wb, ldb, sW1, sW2,
                              C, ldc, sC1, sC2, Cb, bias, Cadd,
                              M, N, K, alpha, zdiv, gelu);
}

static void launch_wt(hipStream_t s, const float* in, __bf16* out,
                      int R, int Ck, int ldo, int nz, long long inStride, long long outStride)
{
  long long total = (long long)Ck * ldo;
  dim3 grid((unsigned)((total + 255) / 256), 1, nz);
  k_wt<<<grid, 256, 0, s>>>(in, out, R, Ck, ldo, inStride, outStride);
}

extern "C" void kernel_launch(void* const* d_in, const int* in_sizes, int n_in,
                              void* d_out, int out_size, void* d_ws, size_t ws_size,
                              hipStream_t stream)
{
  (void)in_sizes; (void)n_in; (void)out_size; (void)ws_size;

  const float* x      = (const float*)d_in[0];
  const float* Wpe    = (const float*)d_in[1];
  const float* bpe    = (const float*)d_in[2];
  const float* cls    = (const float*)d_in[3];
  const float* pos    = (const float*)d_in[4];
  const float* ln1_g  = (const float*)d_in[5];
  const float* ln1_b  = (const float*)d_in[6];
  const float* qkv_w  = (const float*)d_in[7];
  const float* qkv_b  = (const float*)d_in[8];
  const float* pre_w  = (const float*)d_in[9];
  const float* post_w = (const float*)d_in[10];
  const float* proj_w = (const float*)d_in[11];
  const float* proj_b = (const float*)d_in[12];
  const float* ln2_g  = (const float*)d_in[13];
  const float* ln2_b  = (const float*)d_in[14];
  const float* fc1_w  = (const float*)d_in[15];
  const float* fc1_b  = (const float*)d_in[16];
  const float* fc2_w  = (const float*)d_in[17];
  const float* fc2_b  = (const float*)d_in[18];
  const float* lnf_g  = (const float*)d_in[19];
  const float* lnf_b  = (const float*)d_in[20];
  const float* head_w = (const float*)d_in[21];
  const float* head_b = (const float*)d_in[22];
  float* out = (float*)d_out;

  const long long ROWS = (long long)BATCH * NTOK;          // 6304
  const long long PR   = (long long)BATCH * NPATCH;        // 6272

  // ---- workspace arena (float-sized slots; bf16 buffers occupy half) ----
  float* ws = (float*)d_ws;
  float*  t    = ws;                                       // [B,197,768] fp32
  float*  a1   = t  + ROWS * DMODEL;                       // scores fp32 [B,H,197,256]; embed fp32 at start
  float*  a2   = a1 + (long long)BATCH * NHEAD * NMP;      // mix/softmax fp32; gelu bf16 overlay
  float*  nxt  = a2 + (long long)BATCH * NHEAD * NMP;
  __bf16* bigb = (__bf16*)nxt; nxt += (ROWS * 3 * DMODEL) / 2;             // qkv bf16 [6304,2304]
  __bf16* a1b  = (__bf16*)nxt; nxt += ((long long)BATCH * NHEAD * NMP) / 2; // probs bf16
  __bf16* xpb  = (__bf16*)nxt; nxt += (PR * PPE) / 2;                       // patches bf16
  __bf16* hb   = (__bf16*)nxt; nxt += (ROWS * DMODEL) / 2;                  // LN out bf16
  __bf16* obb  = (__bf16*)nxt; nxt += (ROWS * DMODEL) / 2;                  // attn-out bf16
  __bf16* ktb  = (__bf16*)nxt; nxt += ((long long)BATCH * NHEAD * HDIM * NTOKP) / 2; // K^T bf16
  // B-ready bf16 weights (k-major [K][ldo]):
  __bf16* wpeT  = (__bf16*)nxt; nxt += (long long)PPE * DMODEL / 2;
  __bf16* headT = (__bf16*)nxt; nxt += (long long)DMODEL * NCLSP / 2;
  __bf16* qwT   = (__bf16*)nxt; nxt += (long long)DEPTH_ * DMODEL * 3 * DMODEL / 2;
  __bf16* pwT   = (__bf16*)nxt; nxt += (long long)DEPTH_ * DMODEL * DMODEL / 2;
  __bf16* f1T   = (__bf16*)nxt; nxt += (long long)DEPTH_ * DMODEL * MLPD / 2;
  __bf16* f2T   = (__bf16*)nxt; nxt += (long long)DEPTH_ * MLPD * DMODEL / 2;

  float*  emb   = a1;                                      // patch-embed fp32 (pre-attention lifetime)
  __bf16* gelub = (__bf16*)a2;                             // gelu bf16 [6304,3072] overlay

  // ---- one-time (per launch) weight convert + transpose to B-ready bf16 ----
  launch_wt(stream, Wpe,    wpeT,  DMODEL, PPE,    DMODEL, 1, 0, 0);
  launch_wt(stream, head_w, headT, NCLS,   DMODEL, NCLSP,  1, 0, 0);
  launch_wt(stream, qkv_w,  qwT,   3*DMODEL, DMODEL, 3*DMODEL, DEPTH_,
            (long long)3*DMODEL*DMODEL, (long long)DMODEL*3*DMODEL);
  launch_wt(stream, proj_w, pwT,   DMODEL, DMODEL, DMODEL, DEPTH_,
            (long long)DMODEL*DMODEL, (long long)DMODEL*DMODEL);
  launch_wt(stream, fc1_w,  f1T,   MLPD,   DMODEL, MLPD,   DEPTH_,
            (long long)MLPD*DMODEL, (long long)DMODEL*MLPD);
  launch_wt(stream, fc2_w,  f2T,   DMODEL, MLPD,   DMODEL, DEPTH_,
            (long long)DMODEL*MLPD, (long long)MLPD*DMODEL);

  // ---- patch embed ----
  {
    long long n = PR * PPE;
    k_patchify<<<(unsigned)((n + 255) / 256), 256, 0, stream>>>(x, xpb);
    launch_gemm(stream, xpb, PPE, 0, 0, wpeT, DMODEL, 0, 0, emb, DMODEL, 0, 0,
                nullptr, bpe, nullptr, (int)PR, DMODEL, PPE, 1.0f, 1, 1, 0);
    long long n2 = ROWS * DMODEL;
    k_add_cls_pos<<<(unsigned)((n2 + 255) / 256), 256, 0, stream>>>(emb, cls, pos, t);
  }

  // ---- transformer blocks ----
  for (int L = 0; L < DEPTH_; ++L) {
    const float* g1  = ln1_g  + (long long)L * DMODEL;
    const float* b1  = ln1_b  + (long long)L * DMODEL;
    const float* qb  = qkv_b  + (long long)L * 3 * DMODEL;
    const float* wpr = pre_w  + (long long)L * NHEAD * NHEAD;
    const float* wpo = post_w + (long long)L * NHEAD * NHEAD;
    const float* pb  = proj_b + (long long)L * DMODEL;
    const float* g2  = ln2_g  + (long long)L * DMODEL;
    const float* b2  = ln2_b  + (long long)L * DMODEL;
    const float* f1b = fc1_b  + (long long)L * MLPD;
    const float* f2b = fc2_b  + (long long)L * DMODEL;
    const __bf16* qwL = qwT + (long long)L * DMODEL * 3 * DMODEL;
    const __bf16* pwL = pwT + (long long)L * DMODEL * DMODEL;
    const __bf16* f1L = f1T + (long long)L * DMODEL * MLPD;
    const __bf16* f2L = f2T + (long long)L * MLPD * DMODEL;

    // LN1 -> bf16
    k_layernorm<<<(unsigned)ROWS, 256, 0, stream>>>(t, g1, b1, hb);

    // QKV: bf16 only (Q as A operand; V consumed in-place as B operand)
    launch_gemm(stream, hb, DMODEL, 0, 0, qwL, 3 * DMODEL, 0, 0,
                nullptr, 3 * DMODEL, 0, 0, bigb, qb, nullptr,
                (int)ROWS, 3 * DMODEL, DMODEL, 1.0f, 1, 1, 0);

    // K^T -> ktb [B,H,64,256] (zero-padded token dim)
    {
      long long n = (long long)BATCH * NHEAD * HDIM * NTOKP;
      k_kt<<<(unsigned)((n + 255) / 256), 256, 0, stream>>>(bigb, ktb);
    }

    // scores: per (b,h): Q[197,64] x K^T[64,256] -> a1 [B,H,197,(256)]
    launch_gemm(stream,
                bigb, 3 * DMODEL, (long long)NTOK * 3 * DMODEL, HDIM,        // Q bf16
                ktb,  NTOKP,      (long long)NHEAD * HDIM * NTOKP, (long long)HDIM * NTOKP,
                a1,   NTOKP,      (long long)NHEAD * NMP, NMP,
                nullptr, nullptr, nullptr, NTOK, NTOK, HDIM, ATTN_SCALE,
                BATCH * NHEAD, NHEAD, 0);

    // talking heads pre-softmax: a1 -> a2 (fp32)
    {
      long long n = (long long)BATCH * NMP;
      k_mix_heads<<<(unsigned)((n + 255) / 256), 256, 0, stream>>>(a1, wpr, a2, nullptr);
    }
    // softmax in place on a2 (zero-fills padded cols 197..255)
    k_softmax<<<(unsigned)((long long)BATCH * NHEAD * NTOK), 256, 0, stream>>>(a2, NTOK, NTOKP);
    // talking heads post-softmax: a2 -> a1b (bf16, padding stays exactly zero)
    {
      long long n = (long long)BATCH * NMP;
      k_mix_heads<<<(unsigned)((n + 255) / 256), 256, 0, stream>>>(a2, wpo, nullptr, a1b);
    }

    // o = P @ V: per (b,h): [197,256(pad)] x V (k-major in bigb) -> obb bf16
    launch_gemm(stream,
                a1b,               NTOKP,      (long long)NHEAD * NMP,       NMP,  // P bf16
                bigb + 2 * DMODEL, 3 * DMODEL, (long long)NTOK * 3 * DMODEL, HDIM, // V rows=k
                nullptr,           DMODEL,     (long long)NTOK * DMODEL,     HDIM,
                obb, nullptr, nullptr, NTOK, HDIM, NTOKP, 1.0f,
                BATCH * NHEAD, NHEAD, 0);

    // t = t + o @ proj_w^T + proj_b
    launch_gemm(stream, obb, DMODEL, 0, 0, pwL, DMODEL, 0, 0, t, DMODEL, 0, 0,
                nullptr, pb, t, (int)ROWS, DMODEL, DMODEL, 1.0f, 1, 1, 0);

    // LN2 -> bf16
    k_layernorm<<<(unsigned)ROWS, 256, 0, stream>>>(t, g2, b2, hb);

    // fc1 with fused GELU -> bf16 only
    launch_gemm(stream, hb, DMODEL, 0, 0, f1L, MLPD, 0, 0,
                nullptr, MLPD, 0, 0, gelub, f1b, nullptr,
                (int)ROWS, MLPD, DMODEL, 1.0f, 1, 1, 1);

    // t = t + gelu @ fc2_w^T + fc2_b
    launch_gemm(stream, gelub, MLPD, 0, 0, f2L, DMODEL, 0, 0, t, DMODEL, 0, 0,
                nullptr, f2b, t, (int)ROWS, DMODEL, MLPD, 1.0f, 1, 1, 0);
  }

  // ---- final LN + head on cls rows ----
  k_layernorm<<<(unsigned)ROWS, 256, 0, stream>>>(t, lnf_g, lnf_b, hb);
  launch_gemm(stream, hb, NTOK * DMODEL, 0, 0, headT, NCLSP, 0, 0,
              out, NCLS, 0, 0, nullptr, head_b, nullptr,
              BATCH, NCLS, DMODEL, 1.0f, 1, 1, 0);
}
